// MultiHeadAttention_12558484374090
// MI455X (gfx1250) — matrix operand order repaired and checked by reading, not compiled, not run on hardware
//
#include <hip/hip_runtime.h>
#include <stdint.h>

typedef __bf16 bf16_t;
typedef __attribute__((ext_vector_type(16))) __bf16 bf16x16;
typedef __attribute__((ext_vector_type(8)))  __bf16 bf16x8;
typedef __attribute__((ext_vector_type(4)))  __bf16 bf16x4;
typedef __attribute__((ext_vector_type(8)))  float  f32x8;

#define N_EMBD   1024
#define N_HEAD   16
#define HEAD_DIM 64
#define SEQ      2048
#define BATCH    4
#define TOKENS   (BATCH * SEQ)   // 8192
#define NEG_BIG  (-1.0e30f)

// ---- CDNA5 async global->LDS path (guarded; falls back to ld+ds_store) ----
#if defined(__has_builtin)
#  if __has_builtin(__builtin_amdgcn_global_load_async_to_lds_b128) && \
      __has_builtin(__builtin_amdgcn_s_wait_asynccnt)
#    define HAVE_ASYNC_LDS 1
#  endif
#endif
#ifndef HAVE_ASYNC_LDS
#  define HAVE_ASYNC_LDS 0
#endif

// builtin signature (probed from diagnostics): non-const int4* in AS(1)/AS(3)
typedef int v4i_g __attribute__((vector_size(16)));
typedef __attribute__((address_space(1))) v4i_g as1v4i;
typedef __attribute__((address_space(3))) v4i_g as3v4i;

__device__ __forceinline__ void cp16(const bf16_t* __restrict__ g, bf16_t* l) {
#if HAVE_ASYNC_LDS
  __builtin_amdgcn_global_load_async_to_lds_b128(
      (as1v4i*)(uintptr_t)g, (as3v4i*)(uint32_t)(uintptr_t)l, 0, 0);
#else
  *(bf16x8*)l = *(const bf16x8*)g;
#endif
}

template <int N>
__device__ __forceinline__ void wait_async() {
#if HAVE_ASYNC_LDS
  __builtin_amdgcn_s_wait_asynccnt(N);
#endif
}

__device__ __forceinline__ bf16_t f2bf(float f) {
  unsigned u = __builtin_bit_cast(unsigned, f);
  u = (u + 0x7FFFu + ((u >> 16) & 1u)) >> 16;
  unsigned short s = (unsigned short)u;
  return __builtin_bit_cast(bf16_t, s);
}

// ---------------- conversion kernels ----------------

__global__ void cvt_f32_bf16_k(const float* __restrict__ src,
                               bf16_t* __restrict__ dst, int n) {
  int i = (blockIdx.x * blockDim.x + threadIdx.x) * 4;
  if (i + 3 < n) {
    float4 v = *(const float4*)(src + i);
    bf16x4 o;
    o[0] = f2bf(v.x); o[1] = f2bf(v.y); o[2] = f2bf(v.z); o[3] = f2bf(v.w);
    *(bf16x4*)(dst + i) = o;
  }
}

// src [K][N] f32 row-major -> dst [N][K] bf16 row-major (coalesced writes)
__global__ void cvt_transpose_k(const float* __restrict__ src,
                                bf16_t* __restrict__ dst, int K, int N) {
  long idx = (long)blockIdx.x * blockDim.x + threadIdx.x;
  if (idx < (long)K * N) {
    int k = (int)(idx % K);
    int n = (int)(idx / K);
    dst[idx] = f2bf(src[(long)k * N + n]);
  }
}

// ---------------- tiled bf16 WMMA GEMM (double-buffered, async LDS) --------
// C[M][N] = A[M][K] * Bt[N][K]^T + bias
// MODE 0: write f32 C to Cout.   MODE 1: scatter bf16 into Q/K/Vt buffers.
template <int MODE>
__global__ __launch_bounds__(256) void wmma_gemm_k(
    const bf16_t* __restrict__ A, const bf16_t* __restrict__ Bt,
    const float* __restrict__ bias, float* __restrict__ Cout,
    bf16_t* __restrict__ Qb, bf16_t* __restrict__ Kb,
    bf16_t* __restrict__ Vtb, int M, int N, int K) {
  constexpr int LDA = 40;  // padded LDS stride (elements): 80B rows, no conflicts
  __shared__ bf16_t As[2][128 * LDA];
  __shared__ bf16_t Bs[2][128 * LDA];

  const int tid  = threadIdx.x;
  const int lane = tid & 31;
  const int wave = tid >> 5;
  const int wm   = wave >> 2;  // 0..1
  const int wn   = wave & 3;   // 0..3
  const int m0   = blockIdx.y * 128;
  const int n0   = blockIdx.x * 128;

  f32x8 acc[4][2];
#pragma unroll
  for (int mt = 0; mt < 4; ++mt)
#pragma unroll
    for (int nt = 0; nt < 2; ++nt) {
      f32x8 z = {0.f, 0.f, 0.f, 0.f, 0.f, 0.f, 0.f, 0.f};
      acc[mt][nt] = z;
    }

  const int lr   = tid >> 2;        // 0..63
  const int lc   = (tid & 3) << 3;  // 0,8,16,24
  const int arow = lane & 15;
  const int ak0  = (lane >> 4) ? 8 : 0;
  const int bk0  = (lane >> 4) ? 8 : 0;

  auto tile_load = [&](int kk, int buf) {
#pragma unroll
    for (int h = 0; h < 2; ++h) {
      int r = lr + h * 64;
      cp16(&A[(long)(m0 + r) * K + kk + lc], &As[buf][r * LDA + lc]);
      cp16(&Bt[(long)(n0 + r) * K + kk + lc], &Bs[buf][r * LDA + lc]);
    }
  };

  tile_load(0, 0);  // 4 async ops per wave per tile
  const int nk = K >> 5;
  for (int t = 0; t < nk; ++t) {
    const int buf = t & 1;
    if (t + 1 < nk) {
      tile_load((t + 1) << 5, buf ^ 1);  // prefetch next tile
      wait_async<4>();                   // oldest 4 (tile t) retired
    } else {
      wait_async<0>();
    }
    __syncthreads();

    bf16x16 afrag[4];
#pragma unroll
    for (int mt = 0; mt < 4; ++mt) {
      const bf16_t* p = &As[buf][(wm * 64 + mt * 16 + arow) * LDA];
      ((bf16x8*)&afrag[mt])[0] = *(const bf16x8*)(p + ak0);
      ((bf16x8*)&afrag[mt])[1] = *(const bf16x8*)(p + ak0 + 16);
    }
    bf16x16 bfrag[2];
#pragma unroll
    for (int nt = 0; nt < 2; ++nt) {
      const bf16_t* p = &Bs[buf][(wn * 32 + nt * 16 + (lane & 15)) * LDA];
      ((bf16x8*)&bfrag[nt])[0] = *(const bf16x8*)(p + bk0);
      ((bf16x8*)&bfrag[nt])[1] = *(const bf16x8*)(p + bk0 + 16);
    }
#pragma unroll
    for (int mt = 0; mt < 4; ++mt)
#pragma unroll
      for (int nt = 0; nt < 2; ++nt)
        acc[mt][nt] = __builtin_amdgcn_wmma_f32_16x16x32_bf16(
            false, afrag[mt], false, bfrag[nt], (short)0, acc[mt][nt], false, false);
    __syncthreads();
  }

  const int g  = lane >> 4;
  const int cn = lane & 15;
#pragma unroll
  for (int mt = 0; mt < 4; ++mt)
#pragma unroll
    for (int nt = 0; nt < 2; ++nt) {
      int colb = n0 + wn * 32 + nt * 16 + cn;
      float bv = bias[colb];
#pragma unroll
      for (int r = 0; r < 8; ++r) {
        int row = m0 + wm * 64 + mt * 16 + r + 8 * g;
        float v = acc[mt][nt][r] + bv;
        if (MODE == 0) {
          Cout[(long)row * N + colb] = v;
        } else {
          int which = colb >> 10;  // 0=q 1=k 2=v
          int rem = colb & 1023;
          int h = rem >> 6;
          int d = rem & 63;
          int b = row >> 11;
          int t = row & 2047;
          int bh = b * N_HEAD + h;
          bf16_t bx = f2bf(v);
          if (which == 0)      Qb[((long)bh * SEQ + t) * HEAD_DIM + d] = bx;
          else if (which == 1) Kb[((long)bh * SEQ + t) * HEAD_DIM + d] = bx;
          else                 Vtb[((long)bh * HEAD_DIM + d) * SEQ + t] = bx;
        }
      }
    }
}

// ---------------- flash attention (online softmax, WMMA, async KV) --------
// grid (SEQ/64, B*H), block 128 (4 waves, 16 queries each)
__global__ __launch_bounds__(128) void flash_attn_k(
    const bf16_t* __restrict__ Qb, const bf16_t* __restrict__ Kb,
    const bf16_t* __restrict__ Vtb, bf16_t* __restrict__ AO) {
  constexpr int LD = 72;  // padded stride: 144B, 16B-aligned rows
  __shared__ bf16_t Qs[64 * LD];
  __shared__ bf16_t Ks[2][64 * LD];
  __shared__ bf16_t Vs[2][64 * LD];
  __shared__ bf16_t Ps[64 * LD];

  const int tid  = threadIdx.x;
  const int lane = tid & 31;
  const int wave = tid >> 5;
  const int bh   = blockIdx.y;
  const int qt   = blockIdx.x;
  const int q0   = qt * 64;

  const bf16_t* Qg = Qb  + (long)bh * SEQ * HEAD_DIM;
  const bf16_t* Kg = Kb  + (long)bh * SEQ * HEAD_DIM;
  const bf16_t* Vg = Vtb + (long)bh * HEAD_DIM * SEQ;

  auto load_kv = [&](int j, int buf) {  // 8 async ops per wave per tile
    int kv0 = j * 64;
#pragma unroll
    for (int it = 0; it < 4; ++it) {
      int id = tid + it * 128;
      int r = id >> 3, c = (id & 7) << 3;
      cp16(&Kg[(kv0 + r) * HEAD_DIM + c], &Ks[buf][r * LD + c]);
      cp16(&Vg[r * SEQ + kv0 + c], &Vs[buf][r * LD + c]);
    }
  };

#pragma unroll
  for (int it = 0; it < 4; ++it) {  // Q tile: 4 async ops
    int id = tid + it * 128;
    int r = id >> 3, c = (id & 7) << 3;
    cp16(&Qg[(q0 + r) * HEAD_DIM + c], &Qs[r * LD + c]);
  }
  load_kv(0, 0);
  wait_async<8>();  // Q retired (KV0's 8 may remain in flight)
  __syncthreads();

  const int arow = lane & 15;
  const int ak0  = (lane >> 4) ? 8 : 0;
  const int bk0  = (lane >> 4) ? 8 : 0;
  const int g    = lane >> 4;
  const int cn   = lane & 15;

  bf16x16 qf[2];
#pragma unroll
  for (int dc = 0; dc < 2; ++dc) {
    const bf16_t* p = &Qs[(wave * 16 + arow) * LD + dc * 32];
    ((bf16x8*)&qf[dc])[0] = *(const bf16x8*)(p + ak0);
    ((bf16x8*)&qf[dc])[1] = *(const bf16x8*)(p + ak0 + 16);
  }

  f32x8 O[4];
  float m[8], l[8];
#pragma unroll
  for (int nt = 0; nt < 4; ++nt) {
    f32x8 z = {0.f, 0.f, 0.f, 0.f, 0.f, 0.f, 0.f, 0.f};
    O[nt] = z;
  }
#pragma unroll
  for (int r = 0; r < 8; ++r) { m[r] = NEG_BIG; l[r] = 0.f; }

  for (int j = 0; j <= qt; ++j) {
    const int kv0 = j * 64;
    const int buf = j & 1;
    if (j < qt) {
      load_kv(j + 1, buf ^ 1);  // prefetch next KV tile
      wait_async<8>();          // tile j fully resident
    } else {
      wait_async<0>();
    }
    __syncthreads();

    // S = (Q K^T) * scale, causal-masked
    f32x8 S[4];
#pragma unroll
    for (int nt = 0; nt < 4; ++nt) {
      f32x8 z = {0.f, 0.f, 0.f, 0.f, 0.f, 0.f, 0.f, 0.f};
      S[nt] = z;
#pragma unroll
      for (int dc = 0; dc < 2; ++dc) {
        bf16x16 kf;
        const bf16_t* p = &Ks[buf][(nt * 16 + cn) * LD + dc * 32];
        ((bf16x8*)&kf)[0] = *(const bf16x8*)(p + bk0);
        ((bf16x8*)&kf)[1] = *(const bf16x8*)(p + bk0 + 16);
        S[nt] = __builtin_amdgcn_wmma_f32_16x16x32_bf16(
            false, qf[dc], false, kf, (short)0, S[nt], false, false);
      }
    }
    const int qrow0 = q0 + wave * 16 + 8 * g;
#pragma unroll
    for (int nt = 0; nt < 4; ++nt) {
      int kvc = kv0 + nt * 16 + cn;
#pragma unroll
      for (int r = 0; r < 8; ++r) {
        float v = S[nt][r] * 0.125f;  // 1/sqrt(64)
        S[nt][r] = (kvc > qrow0 + r) ? NEG_BIG : v;
      }
    }
    // online softmax; row stats reduced across 16-lane groups stay aligned
    // with the WMMA C layout (no LDS round-trip for stats)
    float mnew[8];
#pragma unroll
    for (int r = 0; r < 8; ++r) {
      float x = fmaxf(fmaxf(S[0][r], S[1][r]), fmaxf(S[2][r], S[3][r]));
#pragma unroll
      for (int off = 8; off >= 1; off >>= 1) x = fmaxf(x, __shfl_xor(x, off, 32));
      mnew[r] = fmaxf(m[r], x);
    }
#pragma unroll
    for (int nt = 0; nt < 4; ++nt)
#pragma unroll
      for (int r = 0; r < 8; ++r) S[nt][r] = __expf(S[nt][r] - mnew[r]);
#pragma unroll
    for (int r = 0; r < 8; ++r) {
      float s = S[0][r] + S[1][r] + S[2][r] + S[3][r];
#pragma unroll
      for (int off = 8; off >= 1; off >>= 1) s += __shfl_xor(s, off, 32);
      float alpha = __expf(m[r] - mnew[r]);
      l[r] = l[r] * alpha + s;
      m[r] = mnew[r];
#pragma unroll
      for (int nt = 0; nt < 4; ++nt) O[nt][r] *= alpha;
    }
    // P: C-layout -> LDS -> A-layout
#pragma unroll
    for (int nt = 0; nt < 4; ++nt)
#pragma unroll
      for (int r = 0; r < 8; ++r)
        Ps[(wave * 16 + r + 8 * g) * LD + nt * 16 + cn] = f2bf(S[nt][r]);
    __syncthreads();

    bf16x16 pf[2];
#pragma unroll
    for (int kc = 0; kc < 2; ++kc) {
      const bf16_t* p = &Ps[(wave * 16 + arow) * LD + kc * 32];
      ((bf16x8*)&pf[kc])[0] = *(const bf16x8*)(p + ak0);
      ((bf16x8*)&pf[kc])[1] = *(const bf16x8*)(p + ak0 + 16);
    }
#pragma unroll
    for (int nt = 0; nt < 4; ++nt)
#pragma unroll
      for (int kc = 0; kc < 2; ++kc) {
        bf16x16 vf;
        const bf16_t* p = &Vs[buf][(nt * 16 + cn) * LD + kc * 32];
        ((bf16x8*)&vf)[0] = *(const bf16x8*)(p + bk0);
        ((bf16x8*)&vf)[1] = *(const bf16x8*)(p + bk0 + 16);
        O[nt] = __builtin_amdgcn_wmma_f32_16x16x32_bf16(
            false, pf[kc], false, vf, (short)0, O[nt], false, false);
      }
    __syncthreads();
  }

  // finalize: out token-major [B*T][C], col = h*64 + d (bf16 for final GEMM)
  const int b = bh >> 4, h = bh & 15;
#pragma unroll
  for (int r = 0; r < 8; ++r) {
    float inv = 1.0f / l[r];
    int q = q0 + wave * 16 + r + 8 * g;
    long row = (long)b * SEQ + q;
#pragma unroll
    for (int nt = 0; nt < 4; ++nt) {
      int col = h * 64 + nt * 16 + cn;
      AO[row * N_EMBD + col] = f2bf(O[nt][r] * inv);
    }
  }
}

// ---------------- launcher ----------------
extern "C" void kernel_launch(void* const* d_in, const int* in_sizes, int n_in,
                              void* d_out, int out_size, void* d_ws, size_t ws_size,
                              hipStream_t stream) {
  const float* x      = (const float*)d_in[0];
  const float* W_attn = (const float*)d_in[1];
  const float* b_attn = (const float*)d_in[2];
  const float* W_proj = (const float*)d_in[3];
  const float* b_proj = (const float*)d_in[4];
  float* out = (float*)d_out;

  char* ws = (char*)d_ws;
  size_t off = 0;
  auto alloc = [&](size_t bytes) {
    char* p = ws + off;
    off = (off + bytes + 255) & ~(size_t)255;
    return p;
  };
  bf16_t* xb  = (bf16_t*)alloc((size_t)TOKENS * N_EMBD * 2);
  bf16_t* wat = (bf16_t*)alloc((size_t)3 * N_EMBD * N_EMBD * 2);
  bf16_t* wpt = (bf16_t*)alloc((size_t)N_EMBD * N_EMBD * 2);
  bf16_t* qb  = (bf16_t*)alloc((size_t)TOKENS * N_EMBD * 2);
  bf16_t* kb  = (bf16_t*)alloc((size_t)TOKENS * N_EMBD * 2);
  bf16_t* vtb = (bf16_t*)alloc((size_t)TOKENS * N_EMBD * 2);
  bf16_t* aob = (bf16_t*)alloc((size_t)TOKENS * N_EMBD * 2);
  (void)ws_size; (void)in_sizes; (void)n_in; (void)out_size;

  int nx = TOKENS * N_EMBD;
  cvt_f32_bf16_k<<<nx / (256 * 4), 256, 0, stream>>>(x, xb, nx);
  cvt_transpose_k<<<(3 * N_EMBD * N_EMBD) / 256, 256, 0, stream>>>(
      W_attn, wat, N_EMBD, 3 * N_EMBD);
  cvt_transpose_k<<<(N_EMBD * N_EMBD) / 256, 256, 0, stream>>>(
      W_proj, wpt, N_EMBD, N_EMBD);

  wmma_gemm_k<1><<<dim3(3 * N_EMBD / 128, TOKENS / 128), 256, 0, stream>>>(
      xb, wat, b_attn, nullptr, qb, kb, vtb, TOKENS, 3 * N_EMBD, N_EMBD);

  flash_attn_k<<<dim3(SEQ / 64, BATCH * N_HEAD), 128, 0, stream>>>(
      qb, kb, vtb, aob);

  wmma_gemm_k<0><<<dim3(N_EMBD / 128, TOKENS / 128), 256, 0, stream>>>(
      aob, wpt, b_proj, out, nullptr, nullptr, nullptr, TOKENS, N_EMBD, N_EMBD);
}
